// CustomMultiInputLSTM_37598143709616
// MI455X (gfx1250) — matrix operand order repaired and checked by reading, not compiled, not run on hardware
//
#include <hip/hip_runtime.h>
#include <hip/hip_bf16.h>
#include <math.h>

// ---------------------------------------------------------------------------
// CustomMultiInputLSTM for MI455X (gfx1250, wave32, WMMA).
//   prep (once/launch): convert 18 recurrence matrices + W_a to transposed
//     bf16 (WMMA B-operand friendly), zero h/c state.
//   per step t (4 stream-ordered kernels, graph-captured):
//     1) gemm_z : Z[18,64,512] = h_bf16 @ UallT  (v_wmma_f32_16x16x32_bf16,
//                 A tile staged in LDS, shared by the block's 8 waves)
//                 + fused tiny input projections (Y@W / x@W_x) + biases
//     2) gates  : i,f,o = sigmoid; l[8,64,512] = tanh(cand)*i  (+bf16 copy)
//     3) gemm_u : u = tanh((l @ W_a) * c + b_a)   (WMMA + LDS A staging)
//     4) update : softmax over 8 channels, c_new=f*c+L, h_new=o*tanh(c_new)
// ---------------------------------------------------------------------------

typedef __attribute__((ext_vector_type(16))) __bf16 bf16x16;
typedef __attribute__((ext_vector_type(8)))  __bf16 bf16x8;
typedef __attribute__((ext_vector_type(8)))  float  v8f;

#define Hn   512
#define Bn   64
#define Tn   1024
#define NG   18            // 4 main gates + 7 aux-i + 7 aux-c
#define BH   (Bn*Hn)       // 32768
#define NTILE1 ((Bn/16)*(NG*Hn/16))   // 4 * 576 = 2304
#define NTILE2 ((8*Bn/16)*(Hn/16))    // 32 * 32 = 1024
#define APITCH 520         // LDS row pitch in bf16 (1040B: 4-bank rotate/row)

__device__ __forceinline__ __bf16 f2bf(float f) {
  union { float f; unsigned u; } v; v.f = f;
  unsigned r = v.u + 0x7FFFu + ((v.u >> 16) & 1u);   // RNE
  union { unsigned short s; __bf16 b; } o; o.s = (unsigned short)(r >> 16);
  return o.b;
}
__device__ __forceinline__ float sigf(float x) { return 1.0f / (1.0f + expf(-x)); }

// Stage a 16x512 bf16 A tile (rows m0..m0+15) into LDS, pitch APITCH.
// 1024 16B chunks, 256 threads -> 4 chunks each, coalesced.
__device__ __forceinline__ void stage_a(const __bf16* __restrict__ src,
                                        int m0, __bf16* As) {
  const int tid = threadIdx.x;
#pragma unroll
  for (int i = 0; i < 4; ++i) {
    const int cid = tid + 256 * i;        // 0..1023
    const int row = cid >> 6;             // 0..15
    const int kc  = (cid & 63) * 8;       // 0..504
    *(bf16x8*)(&As[row * APITCH + kc]) =
        *(const bf16x8*)(src + (size_t)(m0 + row) * Hn + kc);
  }
  __syncthreads();
}

// ------------------------------- prep --------------------------------------
struct PrepP {
  const float* U[NG];       // 18 x [512,512] row-major (k, j)
  const float* Wa;          // [512,512]
  __bf16* uallT;            // [(g*512+j)*512 + k]
  __bf16* waT;              // [j*512 + k]
  float* h; __bf16* hbf; float* c;
};

__global__ __launch_bounds__(256) void k_prep(PrepP p) {
  const size_t N1 = (size_t)NG * Hn * Hn;   // UallT
  const size_t N2 = (size_t)Hn * Hn;        // WaT
  const size_t N3 = BH;                     // state
  size_t stride = (size_t)gridDim.x * blockDim.x;
  for (size_t i = blockIdx.x * (size_t)blockDim.x + threadIdx.x;
       i < N1 + N2 + N3; i += stride) {
    if (i < N1) {
      size_t g = i / (Hn * Hn), rem = i % (Hn * Hn);
      size_t j = rem / Hn, k = rem % Hn;
      p.uallT[(g * Hn + j) * Hn + k] = f2bf(p.U[g][k * Hn + j]);
    } else if (i < N1 + N2) {
      size_t rem = i - N1, j = rem / Hn, k = rem % Hn;
      p.waT[j * Hn + k] = f2bf(p.Wa[k * Hn + j]);
    } else {
      size_t e = i - N1 - N2;
      p.h[e] = 0.0f; p.c[e] = 0.0f; p.hbf[e] = f2bf(0.0f);
    }
  }
}

// ------------------------------ gemm_z -------------------------------------
struct G1P {
  const __bf16* hbf;        // [64,512] row-major bf16
  const __bf16* uallT;      // [(g*512+j), k]
  float* Z;                 // [18,64,512]
  const float* Y;           // [64,8,1024]
  const float* x[7];        // each [64,3,1024]
  const float* Wm[4];       // W_i,W_f,W_c,W_o [8,512]
  const float* bm[4];       // b_i..b_o [512]
  const float* W_i_x;       // [7,3,512]
  const float* b_i_x;       // [7,512]
  const float* W_c_x;       // [7,3,512]
  const float* b_c_x;       // [7,512]
  int t;
};

__global__ __launch_bounds__(256) void k_gemm_z(G1P p) {
  __shared__ __bf16 As[16 * APITCH];                      // 16.25 KB
  const int lane = threadIdx.x & 31;
  const int tile = blockIdx.x * 8 + (threadIdx.x >> 5);   // 2304 tiles, exact
  const int tm = tile / (NG * Hn / 16);                   // 0..3 (same in block)
  const int tn = tile % (NG * Hn / 16);                   // 0..575
  const int m0 = tm * 16;
  const int n0 = tn * 16;                                 // global col 0..9215
  const int g  = n0 >> 9;                                 // uniform per wave
  const int j0 = n0 & (Hn - 1);
  const int half = lane >> 4, l15 = lane & 15;

  stage_a(p.hbf, m0, As);

  const __bf16* arow = As + l15 * APITCH + half * 16;     // LDS
  const __bf16* bcol = p.uallT + (size_t)(n0 + l15) * Hn + half * 16;

  v8f acc = {};
#pragma unroll 4
  for (int kt = 0; kt < Hn / 32; ++kt) {
    const int k0 = kt * 32;
    bf16x8 alo = *(const bf16x8*)(arow + k0);             // ds_load_b128
    bf16x8 ahi = *(const bf16x8*)(arow + k0 + 8);
    bf16x16 a;
#pragma unroll
    for (int i = 0; i < 8; ++i) { a[i] = alo[i]; a[i + 8] = ahi[i]; }
    bf16x16 b = *(const bf16x16*)(bcol + k0);             // 2x global_load_b128
    acc = __builtin_amdgcn_wmma_f32_16x16x32_bf16(
        false, a, false, b, (short)0, acc, false, false);
  }

  // Fused input-projection + bias epilogue, then Z store.
  const int j = j0 + l15;
#pragma unroll
  for (int r = 0; r < 8; ++r) {
    const int b = m0 + r + half * 8;          // batch row 0..63
    float s;
    if (g < 4) {
      const float* W = p.Wm[g];
      const float* y = p.Y + (size_t)b * 8 * Tn + p.t;
      s = p.bm[g][j];
#pragma unroll
      for (int e = 0; e < 8; ++e) s += y[(size_t)e * Tn] * W[e * Hn + j];
    } else if (g < 11) {
      const int k = g - 4;                    // aux input gate: x1 then x2...
      const float* W = p.W_i_x + (size_t)k * 3 * Hn;
      const float* x = (k == 0 ? p.x[0] : p.x[1]) + (size_t)b * 3 * Tn + p.t;
      s = p.b_i_x[k * Hn + j];
#pragma unroll
      for (int cc = 0; cc < 3; ++cc) s += x[(size_t)cc * Tn] * W[cc * Hn + j];
    } else {
      const int k = g - 11;                   // aux candidate: x_{k+1}
      const float* W = p.W_c_x + (size_t)k * 3 * Hn;
      const float* x = p.x[k] + (size_t)b * 3 * Tn + p.t;
      s = p.b_c_x[k * Hn + j];
#pragma unroll
      for (int cc = 0; cc < 3; ++cc) s += x[(size_t)cc * Tn] * W[cc * Hn + j];
    }
    p.Z[(size_t)g * BH + (size_t)b * Hn + j] = acc[r] + s;
  }
}

// ------------------------------ gates --------------------------------------
struct GEP { const float* Z; float* l; __bf16* lbf; float* fb; float* ob; };

__global__ __launch_bounds__(256) void k_gates(GEP p) {
  const int idx = blockIdx.x * 256 + threadIdx.x;   // b*512+j, 0..32767
  const float it = sigf(p.Z[idx]);
  const float ft = sigf(p.Z[1 * BH + idx]);
  const float Ct = tanhf(p.Z[2 * BH + idx]);
  const float ot = sigf(p.Z[3 * BH + idx]);
  p.fb[idx] = ft; p.ob[idx] = ot;
  const float l0 = Ct * it;
  p.l[idx] = l0; p.lbf[idx] = f2bf(l0);
#pragma unroll
  for (int k = 0; k < 7; ++k) {
    const float ix = sigf(p.Z[(4 + k) * BH + idx]);
    const float Cx = tanhf(p.Z[(11 + k) * BH + idx]);
    const float lv = Cx * ix;
    p.l[(k + 1) * BH + idx] = lv;
    p.lbf[(k + 1) * BH + idx] = f2bf(lv);
  }
}

// ------------------------------ gemm_u -------------------------------------
struct G2P {
  const __bf16* lbf;        // [8*64, 512] row-major (row = k*64+b)
  const __bf16* waT;        // [col, k]
  const float* c;           // [64,512]
  const float* b_a;         // [512]
  float* u;                 // [8*64, 512]
};

__global__ __launch_bounds__(256) void k_gemm_u(G2P p) {
  __shared__ __bf16 As[16 * APITCH];
  const int lane = threadIdx.x & 31;
  const int tile = blockIdx.x * 8 + (threadIdx.x >> 5);   // 1024 tiles, exact
  const int tm = tile >> 5, tn = tile & 31;               // tm same in block
  const int m0 = tm * 16, n0 = tn * 16;
  const int half = lane >> 4, l15 = lane & 15;

  stage_a(p.lbf, m0, As);

  const __bf16* arow = As + l15 * APITCH + half * 16;     // LDS
  const __bf16* bcol = p.waT + (size_t)(n0 + l15) * Hn + half * 16;

  v8f acc = {};
#pragma unroll 4
  for (int kt = 0; kt < Hn / 32; ++kt) {
    const int k0 = kt * 32;
    bf16x8 alo = *(const bf16x8*)(arow + k0);
    bf16x8 ahi = *(const bf16x8*)(arow + k0 + 8);
    bf16x16 a;
#pragma unroll
    for (int i = 0; i < 8; ++i) { a[i] = alo[i]; a[i + 8] = ahi[i]; }
    bf16x16 b = *(const bf16x16*)(bcol + k0);
    acc = __builtin_amdgcn_wmma_f32_16x16x32_bf16(
        false, a, false, b, (short)0, acc, false, false);
  }

  const int j = n0 + l15;
#pragma unroll
  for (int r = 0; r < 8; ++r) {
    const int row = m0 + r + half * 8;        // k*64 + b
    const int b = row & (Bn - 1);
    const float uu = tanhf(acc[r] * p.c[(size_t)b * Hn + j] + p.b_a[j]);
    p.u[(size_t)row * Hn + j] = uu;
  }
}

// ------------------------------ update -------------------------------------
struct UP {
  const float* u;           // [8,64,512]
  const float* l;           // [8,64,512]
  const float* fb; const float* ob;
  float* c; float* h; __bf16* hbf;
  float* out;               // [h_T | hidden_seq[B,T,H]]
  int t;
};

__global__ __launch_bounds__(256) void k_update(UP p) {
  const int idx = blockIdx.x * 256 + threadIdx.x;   // b*512+j
  const int b = idx >> 9, j = idx & (Hn - 1);
  float uv[8], um = -INFINITY;
#pragma unroll
  for (int k = 0; k < 8; ++k) { uv[k] = p.u[k * BH + idx]; um = fmaxf(um, uv[k]); }
  float se = 0.0f, Ls = 0.0f;
#pragma unroll
  for (int k = 0; k < 8; ++k) {
    const float e = expf(uv[k] - um);
    se += e;
    Ls += e * p.l[k * BH + idx];
  }
  const float L = Ls / se;
  const float cn = p.fb[idx] * p.c[idx] + L;
  const float hn = p.ob[idx] * tanhf(cn);
  p.c[idx] = cn; p.h[idx] = hn; p.hbf[idx] = f2bf(hn);
  p.out[(size_t)BH + ((size_t)b * Tn + p.t) * Hn + j] = hn;   // hidden_seq
  if (p.t == Tn - 1) p.out[idx] = hn;                         // h_T
}

// ---------------------------------------------------------------------------
extern "C" void kernel_launch(void* const* d_in, const int* in_sizes, int n_in,
                              void* d_out, int out_size, void* d_ws, size_t ws_size,
                              hipStream_t stream) {
  (void)in_sizes; (void)n_in; (void)out_size; (void)ws_size;
  const float* Y   = (const float*)d_in[0];
  const float* xs[7];
  for (int i = 0; i < 7; ++i) xs[i] = (const float*)d_in[1 + i];
  const float* W_i = (const float*)d_in[8],  *U_i = (const float*)d_in[9],  *b_i = (const float*)d_in[10];
  const float* W_f = (const float*)d_in[11], *U_f = (const float*)d_in[12], *b_f = (const float*)d_in[13];
  const float* W_c = (const float*)d_in[14], *U_c = (const float*)d_in[15], *b_c = (const float*)d_in[16];
  const float* W_o = (const float*)d_in[17], *U_o = (const float*)d_in[18], *b_o = (const float*)d_in[19];
  const float* W_i_x = (const float*)d_in[20], *U_i_x = (const float*)d_in[21], *b_i_x = (const float*)d_in[22];
  const float* W_c_x = (const float*)d_in[23], *U_c_x = (const float*)d_in[24], *b_c_x = (const float*)d_in[25];
  const float* W_a = (const float*)d_in[26], *b_a = (const float*)d_in[27];
  float* out = (float*)d_out;

  // Workspace carve-up (all 256B-aligned).
  char* w = (char*)d_ws;
  __bf16* uallT = (__bf16*)(w);                         // 18*512*512*2 = 9437184
  __bf16* waT   = (__bf16*)(w + 9437184);               // 524288
  float*  h     = (float*) (w + 9961472);               // 131072
  __bf16* hbf   = (__bf16*)(w + 10092544);              // 65536
  float*  c     = (float*) (w + 10158080);              // 131072
  float*  Z     = (float*) (w + 10289152);              // 2359296
  float*  l     = (float*) (w + 12648448);              // 1048576
  __bf16* lbf   = (__bf16*)(w + 13697024);              // 524288
  float*  fb    = (float*) (w + 14221312);              // 131072
  float*  ob    = (float*) (w + 14352384);              // 131072
  float*  u     = (float*) (w + 14483456);              // 1048576  (end ~14.8MB)

  // One-time weight transpose/convert + state init.
  PrepP pp;
  pp.U[0] = U_i; pp.U[1] = U_f; pp.U[2] = U_c; pp.U[3] = U_o;
  for (int k = 0; k < 7; ++k) {
    pp.U[4 + k]  = U_i_x + (size_t)k * Hn * Hn;
    pp.U[11 + k] = U_c_x + (size_t)k * Hn * Hn;
  }
  pp.Wa = W_a; pp.uallT = uallT; pp.waT = waT; pp.h = h; pp.hbf = hbf; pp.c = c;
  hipLaunchKernelGGL(k_prep, dim3(2048), dim3(256), 0, stream, pp);

  G1P g1;
  g1.hbf = hbf; g1.uallT = uallT; g1.Z = Z; g1.Y = Y;
  for (int i = 0; i < 7; ++i) g1.x[i] = xs[i];
  g1.Wm[0] = W_i; g1.Wm[1] = W_f; g1.Wm[2] = W_c; g1.Wm[3] = W_o;
  g1.bm[0] = b_i; g1.bm[1] = b_f; g1.bm[2] = b_c; g1.bm[3] = b_o;
  g1.W_i_x = W_i_x; g1.b_i_x = b_i_x; g1.W_c_x = W_c_x; g1.b_c_x = b_c_x;

  GEP ge; ge.Z = Z; ge.l = l; ge.lbf = lbf; ge.fb = fb; ge.ob = ob;
  G2P g2; g2.lbf = lbf; g2.waT = waT; g2.c = c; g2.b_a = b_a; g2.u = u;
  UP  up; up.u = u; up.l = l; up.fb = fb; up.ob = ob;
  up.c = c; up.h = h; up.hbf = hbf; up.out = out;

  for (int t = 0; t < Tn; ++t) {
    g1.t = t; up.t = t;
    hipLaunchKernelGGL(k_gemm_z, dim3(NTILE1 / 8), dim3(256), 0, stream, g1);
    hipLaunchKernelGGL(k_gates,  dim3(BH / 256),   dim3(256), 0, stream, ge);
    hipLaunchKernelGGL(k_gemm_u, dim3(NTILE2 / 8), dim3(256), 0, stream, g2);
    hipLaunchKernelGGL(k_update, dim3(BH / 256),   dim3(256), 0, stream, up);
  }
}
